// SimpleGPT2_1898375544945
// MI455X (gfx1250) — matrix operand order repaired and checked by reading, not compiled, not run on hardware
//
#include <hip/hip_runtime.h>
#include <stdint.h>

#define B_ 4
#define S_ 1024
#define D_ 768
#define H_ 12
#define L_ 12
#define V_ 50257
#define HD_ 64
#define EPS_ 1e-5f

typedef __attribute__((ext_vector_type(16))) __bf16 v16bf;
typedef __attribute__((ext_vector_type(8)))  float v8f;

// Convert+pack two fp32 into one dword of bf16 (native cvt if available).
__device__ __forceinline__ uint32_t pack_bf2(float lo, float hi) {
  union { __bf16 b[2]; uint32_t u; } t;
  t.b[0] = (__bf16)lo;
  t.b[1] = (__bf16)hi;
  return t.u;
}

__device__ __forceinline__ v8f wmma_bf16(v16bf a, v16bf b, v8f c) {
  return __builtin_amdgcn_wmma_f32_16x16x32_bf16(
      /*neg_a=*/false, a, /*neg_b=*/false, b,
      /*c_mod=*/(short)0, c, /*reuse_a=*/false, /*reuse_b=*/false);
}

// A fragment: 16x32 (MxK) bf16 in contiguous k order (not the ISA table order), read as 8 dword pairs.
// lane&15 = row; lane half selects K subgroup; vgpr v holds K pair at
// k = half*16 + 2*v.
__device__ __forceinline__ v16bf load_frag_a_lds(const __bf16* base, int ldk) {
  int lane = threadIdx.x & 31;
  int row  = lane & 15;
  int half = (lane >> 4) & 1;
  const uint32_t* p = (const uint32_t*)(base + row * ldk);  // ldk even, base 4B-aligned
  union { uint32_t u[8]; v16bf v; } f;
#pragma unroll
  for (int vr = 0; vr < 8; ++vr) {
    int k = half * 16 + 2 * vr;
    f.u[vr] = p[k >> 1];
  }
  return f.v;
}

// B fragment: 32x16 (KxN) bf16 from LDS stored transposed [n][k]:
// lane&15 = col n; vgpr v holds K pair k = half*16 + 2v (8 contiguous dwords).
__device__ __forceinline__ v16bf load_frag_b_lds(const __bf16* base, int ldk) {
  int lane = threadIdx.x & 31;
  int n    = lane & 15;
  int half = (lane >> 4) & 1;
  const uint32_t* p = (const uint32_t*)(base + n * ldk + half * 16);
  union { uint32_t u[8]; v16bf v; } f;
#pragma unroll
  for (int vr = 0; vr < 8; ++vr) f.u[vr] = p[vr];
  return f.v;
}

// ---------------- embedding ----------------
__global__ void embed_kernel(const int* __restrict__ ids, const float* __restrict__ wte,
                             const float* __restrict__ wpe, float* __restrict__ h) {
  int i = blockIdx.x * blockDim.x + threadIdx.x;
  if (i >= B_ * S_ * D_) return;
  int d  = i % D_;
  int bs = i / D_;
  int s  = bs % S_;
  int tok = ids[bs];
  h[i] = wte[(size_t)tok * D_ + d] + wpe[(size_t)s * D_ + d];
}

// ---------------- layernorm (one block per row, D = 3*256) ----------------
__global__ __launch_bounds__(256) void ln_kernel(const float* __restrict__ in,
                                                 const float* __restrict__ w,
                                                 const float* __restrict__ bvec,
                                                 float* __restrict__ out) {
  __shared__ float r1[256], r2[256];
  int row = blockIdx.x, tid = threadIdx.x;
  const float* x = in + (size_t)row * D_;
  float a0 = x[tid], a1 = x[tid + 256], a2 = x[tid + 512];
  r1[tid] = a0 + a1 + a2;
  r2[tid] = a0 * a0 + a1 * a1 + a2 * a2;
  __syncthreads();
  for (int st = 128; st > 0; st >>= 1) {
    if (tid < st) { r1[tid] += r1[tid + st]; r2[tid] += r2[tid + st]; }
    __syncthreads();
  }
  float mean = r1[0] * (1.0f / D_);
  float var  = r2[0] * (1.0f / D_) - mean * mean;
  float inv  = rsqrtf(var + EPS_);
  float* o = out + (size_t)row * D_;
  o[tid]       = (a0 - mean) * inv * w[tid]       + bvec[tid];
  o[tid + 256] = (a1 - mean) * inv * w[tid + 256] + bvec[tid + 256];
  o[tid + 512] = (a2 - mean) * inv * w[tid + 512] + bvec[tid + 512];
}

// ---------------- bf16 WMMA GEMM: C = act(A@W + bias) (+resid) ----------------
// 128x128 block tile, 8 waves of 32x64 (2x4 fragments, 8 WMMA/wave/K-step),
// double-buffered LDS staging (1 barrier per K-step).
#define BM 128
#define BN 128
#define BK 32
#define LDPAD 34

template <int ACT, bool HAS_BIAS, bool RESID>
__global__ __launch_bounds__(256) void gemm_kernel(
    const float* __restrict__ A, const float* __restrict__ W,
    const float* __restrict__ bias, const float* __restrict__ resid,
    float* __restrict__ C, int M, int N, int K) {
  __shared__ __bf16 As[2][BM][LDPAD];
  __shared__ __bf16 Bs[2][BN][LDPAD];
  const int tid  = threadIdx.x;
  const int wid  = tid >> 5;
  const int wr   = wid >> 1, wc = wid & 1;  // 4x2 waves of 32x64 tiles
  const int lane = tid & 31, half = lane >> 4, ln = lane & 15;
  const int mbase = blockIdx.y * BM;
  const int nbase = blockIdx.x * BN;

  auto stage = [&](int bufi, int kk0) {
    // A tile 128x32: pack fp32 pairs -> bf16 dwords (2048 dwords, 8/thread)
#pragma unroll
    for (int i = 0; i < (BM * BK / 2) / 256; ++i) {
      int f = i * 256 + tid;
      int r = f >> 4, cp = f & 15;
      const float* src = &A[(size_t)(mbase + r) * K + kk0 + 2 * cp];
      *(uint32_t*)&As[bufi][r][2 * cp] = pack_bf2(src[0], src[1]);
    }
    // B tile transposed: W[K,N] -> Bs[n][k] (2048 dwords, 8/thread)
#pragma unroll
    for (int i = 0; i < (BN * BK / 2) / 256; ++i) {
      int f = i * 256 + tid;
      int n = f & (BN - 1), kp = f >> 7;  // kp: 0..15
      int gn = nbase + n;
      float lo = (gn < N) ? W[(size_t)(kk0 + 2 * kp) * N + gn] : 0.0f;
      float hi = (gn < N) ? W[(size_t)(kk0 + 2 * kp + 1) * N + gn] : 0.0f;
      *(uint32_t*)&Bs[bufi][n][2 * kp] = pack_bf2(lo, hi);
    }
  };

  v8f zero = {};
  v8f acc[2][4];
#pragma unroll
  for (int i = 0; i < 2; ++i)
#pragma unroll
    for (int j = 0; j < 4; ++j) acc[i][j] = zero;

  stage(0, 0);
  int buf = 0;
  for (int k0 = 0; k0 < K; k0 += BK) {
    __syncthreads();  // buf staged for all; previous buf^1 readers done
    if (k0 + BK < K) stage(buf ^ 1, k0 + BK);
    // prefetch two K-tiles ahead (global_prefetch_b8 hint)
    if (k0 + 2 * BK < K) {
      __builtin_prefetch(&A[(size_t)(mbase + (tid >> 1)) * K + k0 + 2 * BK + (tid & 1) * 16], 0, 3);
      int gn = nbase + (tid & 7) * 16;
      if (gn < N)
        __builtin_prefetch(&W[(size_t)(k0 + 2 * BK + (tid >> 3)) * N + gn], 0, 3);
    }
    v16bf af[2], bfv[4];
#pragma unroll
    for (int mi = 0; mi < 2; ++mi)
      af[mi] = load_frag_a_lds(&As[buf][wr * 32 + mi * 16][0], LDPAD);
#pragma unroll
    for (int ni = 0; ni < 4; ++ni)
      bfv[ni] = load_frag_b_lds(&Bs[buf][wc * 64 + ni * 16][0], LDPAD);
#pragma unroll
    for (int mi = 0; mi < 2; ++mi)
#pragma unroll
      for (int ni = 0; ni < 4; ++ni)
        acc[mi][ni] = wmma_bf16(af[mi], bfv[ni], acc[mi][ni]);
    buf ^= 1;
  }

  // epilogue: C layout row = r + half*8, col = lane&15
#pragma unroll
  for (int mi = 0; mi < 2; ++mi)
#pragma unroll
    for (int ni = 0; ni < 4; ++ni)
#pragma unroll
      for (int r = 0; r < 8; ++r) {
        int m = mbase + wr * 32 + mi * 16 + r + half * 8;
        int n = nbase + wc * 64 + ni * 16 + ln;
        if (n < N) {
          float v = acc[mi][ni][r];
          if (HAS_BIAS) v += bias[n];
          if (ACT == 1) v = 0.5f * v * (1.0f + erff(v * 0.70710678118f));  // exact GELU
          if (RESID) v += resid[(size_t)m * N + n];
          C[(size_t)m * N + n] = v;
        }
      }
}

// ---------------- flash attention (no causal mask) ----------------
// One block = 8 waves over one (b,h); each wave owns 16 query rows.
// K/V tiles double-buffered in LDS per 32-key online-softmax step.
__global__ __launch_bounds__(256) void attn_kernel(const float* __restrict__ qkv,
                                                   float* __restrict__ y) {
  __shared__ __bf16 Qs[128][66];        // [qrow][hd]
  __shared__ __bf16 Ks[2][32][66];      // [key][hd]   (B-frag: n=key, k=hd)
  __shared__ __bf16 Vs[2][64][34];      // [hd][key]   (B-frag: n=hd,  k=key)
  __shared__ __bf16 Plds[8][16][34];    // per-wave P tile (C->A layout transpose)
  const int tid = threadIdx.x, wid = tid >> 5;
  const int lane = tid & 31, half = lane >> 4, ln = lane & 15;
  const int b = blockIdx.z, h = blockIdx.y;
  const int qblk = blockIdx.x * 128;
  const size_t bS = (size_t)b * S_;
  const int qoff = h * HD_;
  const int koff = D_ + h * HD_;
  const int voff = 2 * D_ + h * HD_;
  const float scale = 0.125f;  // 1/sqrt(64)

  auto stage_kv = [&](int bufi, int kbase) {
    // K tile 32x64 (1024 dwords, 4/thread), coalesced over hd
#pragma unroll
    for (int i = 0; i < 4; ++i) {
      int f = i * 256 + tid;
      int key = f >> 5, hp = f & 31;
      const float* src = &qkv[(bS + kbase + key) * (size_t)(3 * D_) + koff + 2 * hp];
      *(uint32_t*)&Ks[bufi][key][2 * hp] = pack_bf2(src[0], src[1]);
    }
    // V tile transposed [hd][key] (1024 dwords, 4/thread), coalesced over hd
#pragma unroll
    for (int i = 0; i < 4; ++i) {
      int f = i * 256 + tid;
      int hd = f & 63, kp = f >> 6;  // kp: 0..15
      float lo = qkv[(bS + kbase + 2 * kp)     * (size_t)(3 * D_) + voff + hd];
      float hi = qkv[(bS + kbase + 2 * kp + 1) * (size_t)(3 * D_) + voff + hd];
      *(uint32_t*)&Vs[bufi][hd][2 * kp] = pack_bf2(lo, hi);
    }
  };

  // stage Q tile 128x64 (coalesced) + first K/V tile
#pragma unroll
  for (int i = 0; i < 16; ++i) {
    int f = i * 256 + tid;  // 0..4095
    int row = f >> 5, hp = f & 31;
    const float* src = &qkv[(bS + qblk + row) * (size_t)(3 * D_) + qoff + 2 * hp];
    *(uint32_t*)&Qs[row][2 * hp] = pack_bf2(src[0], src[1]);
  }
  stage_kv(0, 0);
  __syncthreads();
  v16bf aq[2];
  aq[0] = load_frag_a_lds(&Qs[wid * 16][0], 66);
  aq[1] = load_frag_a_lds(&Qs[wid * 16][32], 66);

  v8f zero = {};
  v8f o[4];
#pragma unroll
  for (int ni = 0; ni < 4; ++ni) o[ni] = zero;
  float mrow[8], lrow[8];
#pragma unroll
  for (int r = 0; r < 8; ++r) { mrow[r] = -3.0e38f; lrow[r] = 0.0f; }

  int buf = 0;
  for (int kb = 0; kb < S_; kb += 32) {
    __syncthreads();  // buf staged for all; buf^1 readers from last step done
    if (kb + 32 < S_) stage_kv(buf ^ 1, kb + 32);

    // scores 16x32: hoist all four K fragments into distinct registers so the
    // WMMAs can overlap the DS loads (partial dscnt waits instead of 0)
    v16bf bk00 = load_frag_b_lds(&Ks[buf][0][0],   66);
    v16bf bk10 = load_frag_b_lds(&Ks[buf][16][0],  66);
    v16bf bk01 = load_frag_b_lds(&Ks[buf][0][32],  66);
    v16bf bk11 = load_frag_b_lds(&Ks[buf][16][32], 66);
    v8f s0 = zero, s1 = zero;
    s0 = wmma_bf16(aq[0], bk00, s0);
    s1 = wmma_bf16(aq[0], bk10, s1);
    s0 = wmma_bf16(aq[1], bk01, s0);
    s1 = wmma_bf16(aq[1], bk11, s1);

    // online softmax; row stats replicated across the 16 lanes of a half
    float p0[8], p1[8], corr[8];
#pragma unroll
    for (int r = 0; r < 8; ++r) {
      float x = fmaxf(s0[r], s1[r]) * scale;
#pragma unroll
      for (int off = 1; off < 16; off <<= 1) x = fmaxf(x, __shfl_xor(x, off, 32));
      float mn = fmaxf(mrow[r], x);
      corr[r] = __expf(mrow[r] - mn);
      mrow[r] = mn;
      p0[r] = __expf(s0[r] * scale - mn);
      p1[r] = __expf(s1[r] * scale - mn);
      float rs = p0[r] + p1[r];
#pragma unroll
      for (int off = 1; off < 16; off <<= 1) rs += __shfl_xor(rs, off, 32);
      lrow[r] = lrow[r] * corr[r] + rs;
    }
#pragma unroll
    for (int ni = 0; ni < 4; ++ni)
#pragma unroll
      for (int r = 0; r < 8; ++r) o[ni][r] *= corr[r];

    // P tile: C layout -> LDS -> A fragment (same-wave DS ops stay in order)
#pragma unroll
    for (int r = 0; r < 8; ++r) {
      Plds[wid][r + half * 8][ln]      = (__bf16)p0[r];
      Plds[wid][r + half * 8][16 + ln] = (__bf16)p1[r];
    }
    v16bf pa = load_frag_a_lds(&Plds[wid][0][0], 34);

    // o += P @ V : hoist all four V fragments, then issue the WMMAs
    v16bf bv[4];
#pragma unroll
    for (int ni = 0; ni < 4; ++ni)
      bv[ni] = load_frag_b_lds(&Vs[buf][ni * 16][0], 34);
#pragma unroll
    for (int ni = 0; ni < 4; ++ni)
      o[ni] = wmma_bf16(pa, bv[ni], o[ni]);
    buf ^= 1;
  }

#pragma unroll
  for (int ni = 0; ni < 4; ++ni)
#pragma unroll
    for (int r = 0; r < 8; ++r) {
      int q = qblk + wid * 16 + r + half * 8;
      y[(bS + q) * (size_t)D_ + qoff + ni * 16 + ln] = o[ni][r] / lrow[r];
    }
}

// ---------------- host orchestration ----------------
extern "C" void kernel_launch(void* const* d_in, const int* in_sizes, int n_in,
                              void* d_out, int out_size, void* d_ws, size_t ws_size,
                              hipStream_t stream) {
  (void)in_sizes; (void)n_in; (void)out_size; (void)ws_size;
  const int*   ids   = (const int*)d_in[0];
  const float* wte   = (const float*)d_in[1];
  const float* wpe   = (const float*)d_in[2];
  const float* ln1w  = (const float*)d_in[3];
  const float* ln1b  = (const float*)d_in[4];
  const float* attnw = (const float*)d_in[5];
  const float* attnb = (const float*)d_in[6];
  const float* atpw  = (const float*)d_in[7];
  const float* atpb  = (const float*)d_in[8];
  const float* ln2w  = (const float*)d_in[9];
  const float* ln2b  = (const float*)d_in[10];
  const float* fcw   = (const float*)d_in[11];
  const float* fcb   = (const float*)d_in[12];
  const float* pjw   = (const float*)d_in[13];
  const float* pjb   = (const float*)d_in[14];
  const float* lnfw  = (const float*)d_in[15];
  const float* lnfb  = (const float*)d_in[16];
  const float* lmw   = (const float*)d_in[17];
  float* out = (float*)d_out;

  const int M = B_ * S_;  // 4096
  float* h   = (float*)d_ws;               // [M, D]
  float* x   = h   + (size_t)M * D_;       // [M, D]
  float* qkv = x   + (size_t)M * D_;       // [M, 3D]
  float* yb  = qkv + (size_t)M * 3 * D_;   // [M, D]
  float* fc  = yb  + (size_t)M * D_;       // [M, 4D]

  embed_kernel<<<(M * D_ + 255) / 256, 256, 0, stream>>>(ids, wte, wpe, h);

  for (int l = 0; l < L_; ++l) {
    ln_kernel<<<M, 256, 0, stream>>>(h, ln1w + l * D_, ln1b + l * D_, x);
    gemm_kernel<0, true, false><<<dim3((3 * D_ + BN - 1) / BN, M / BM), 256, 0, stream>>>(
        x, attnw + (size_t)l * D_ * 3 * D_, attnb + l * 3 * D_, nullptr, qkv, M, 3 * D_, D_);
    attn_kernel<<<dim3(S_ / 128, H_, B_), 256, 0, stream>>>(qkv, yb);
    gemm_kernel<0, true, true><<<dim3((D_ + BN - 1) / BN, M / BM), 256, 0, stream>>>(
        yb, atpw + (size_t)l * D_ * D_, atpb + l * D_, h, h, M, D_, D_);
    ln_kernel<<<M, 256, 0, stream>>>(h, ln2w + l * D_, ln2b + l * D_, x);
    gemm_kernel<1, true, false><<<dim3((4 * D_ + BN - 1) / BN, M / BM), 256, 0, stream>>>(
        x, fcw + (size_t)l * D_ * 4 * D_, fcb + l * 4 * D_, nullptr, fc, M, 4 * D_, D_);
    gemm_kernel<0, true, true><<<dim3((D_ + BN - 1) / BN, M / BM), 256, 0, stream>>>(
        fc, pjw + (size_t)l * 4 * D_ * D_, pjb + l * D_, h, h, M, D_, 4 * D_);
  }
  ln_kernel<<<M, 256, 0, stream>>>(h, lnfw, lnfb, x);
  gemm_kernel<0, false, false><<<dim3((V_ + BN - 1) / BN, M / BM), 256, 0, stream>>>(
      x, lmw, nullptr, nullptr, out, M, V_, D_);
}
